// ObjectDetector_64158221468308
// MI455X (gfx1250) — matrix operand order repaired and checked by reading, not compiled, not run on hardware
//
#include <hip/hip_runtime.h>
#include <hip/hip_bf16.h>
#include <stdint.h>

// ---------------- types ----------------
typedef __attribute__((ext_vector_type(16))) __bf16 v16bf;
typedef __attribute__((ext_vector_type(8)))  __bf16 v8bf;
typedef __attribute__((ext_vector_type(8)))  float  v8f;

#define DEV __device__ __forceinline__

// ---------------- config ----------------
#define NIMG      2
#define IMGF      224.0f
#define FHH       7
#define NS        49          // 7*7
#define CFEAT     1024
#define AANCH     15
#define NANCH     735         // 49*15
#define NCLS      37
#define NROI      1470        // NIMG*NANCH
#define KPOOL     50176       // 1024*49
#define KRPN      9216        // 1024*9
#define CANDK     2048
#define NALL      26460       // 735*36
#define NSORT     32768
#define BBOX_CLIP_F 4.135166556742356f   // ln(1000/16)

// =====================================================================
// Generic WMMA BF16 GEMM:  C[M,N] = act( A[M,K] * B[N,K]^T + bias[N] )
// A: AT (float or __bf16) row-major lda; B: float row-major [N,K].
// block = 128 threads (4 waves), tile 64x64, K-step 32.
// K must be a multiple of 32 (true for all call sites: 1024/9216/50176).
// Out-of-range tile rows are address-clamped, not zeroed: A-row r only
// feeds C-row r and B-row n only feeds C-col n, both masked at the store.
// ASYNC_A: A is bf16 -> stage A tiles with global_load_async_to_lds_b128
// (ASYNCcnt-tracked DMA into LDS), no VGPR round-trip.
// =====================================================================
template<typename AT, typename OT, bool RELU, bool ASYNC_A>
__global__ __launch_bounds__(128)
void gemm_wmma_bf16(const AT* __restrict__ A, int lda,
                    const float* __restrict__ Bw,
                    const float* __restrict__ bias,
                    OT* __restrict__ C, int ldc,
                    int M, int N, int K)
{
    static_assert(!ASYNC_A || sizeof(AT) == 2, "async A path requires bf16 A");
    constexpr int LDP = 40;  // padded LDS pitch (bf16 elems) to dodge bank conflicts
    __shared__ __align__(16) __bf16 As[64 * LDP];
    __shared__ __align__(16) __bf16 Bs[64 * LDP];

    const int tid  = threadIdx.x;
    const int wid  = tid >> 5;         // wave32
    const int lane = tid & 31;
    const int wr   = wid >> 1, wc = wid & 1;
    const int lm   = lane & 15, half = lane >> 4;

    const int m0 = blockIdx.y * 64;
    const int n0 = blockIdx.x * 64;

    v8f acc[2][2] = {};

    const int r0 = tid >> 2;           // 0..31
    const int cg = (tid & 3) * 8;      // 0,8,16,24

    union Frag { v16bf v; v8bf h8[2]; };

    for (int k0 = 0; k0 < K; k0 += 32) {
        // ---- stage A and B 64x32 tiles into LDS as bf16 (branch-free) ----
        #pragma unroll
        for (int rr = 0; rr < 2; ++rr) {
            const int r = r0 + rr * 32;
            // A tile
            {
                const int mc = min(m0 + r, M - 1);     // clamp: masked at store
                const AT* src = A + (size_t)mc * lda + k0 + cg;
                if constexpr (ASYNC_A) {
                    unsigned ldsa = (unsigned)(size_t)&As[r * LDP + cg];
                    unsigned long long ga = (unsigned long long)(size_t)src;
                    asm volatile("global_load_async_to_lds_b128 %0, %1, off"
                                 :: "v"(ldsa), "v"(ga) : "memory");
                } else {
                    v8bf tmp;
                    #pragma unroll
                    for (int j = 0; j < 8; ++j) tmp[j] = (__bf16)(float)src[j];
                    *(v8bf*)&As[r * LDP + cg] = tmp;
                    if (k0 + 32 < K)
                        __builtin_prefetch((const void*)(src + 32), 0, 1);
                }
            }
            // B tile (weights [N,K], f32 -> bf16)
            {
                const int nc = min(n0 + r, N - 1);     // clamp: masked at store
                const float* src = Bw + (size_t)nc * K + k0 + cg;
                v8bf tmp;
                #pragma unroll
                for (int j = 0; j < 8; ++j) tmp[j] = (__bf16)src[j];
                *(v8bf*)&Bs[r * LDP + cg] = tmp;
                if (k0 + 32 < K)
                    __builtin_prefetch((const void*)(src + 32), 0, 1);
            }
        }
        if constexpr (ASYNC_A)
            asm volatile("s_wait_asynccnt 0x0" ::: "memory");
        __syncthreads();

        // ---- load fragments per ISA register layouts and issue WMMA ----
        Frag aF[2], bF[2];
        #pragma unroll
        for (int t = 0; t < 2; ++t) {
            const int ra = wr * 32 + t * 16 + lm;   // A row (M)
            // A lane layout: elems 0..7 -> K = half*8.. ; 8..15 -> K = 16+half*8..
            aF[t].h8[0] = *(const v8bf*)&As[ra * LDP + half * 8];
            aF[t].h8[1] = *(const v8bf*)&As[ra * LDP + 16 + half * 8];
            const int rb = wc * 32 + t * 16 + lm;   // B column (N) = weight row
            // B lane layout: elems 0..7 -> K = half*8.. ; 8..15 -> K = 16+half*8..
            bF[t].h8[0] = *(const v8bf*)&Bs[rb * LDP + half * 8];
            bF[t].h8[1] = *(const v8bf*)&Bs[rb * LDP + 16 + half * 8];
        }
        #pragma unroll
        for (int mt = 0; mt < 2; ++mt)
            #pragma unroll
            for (int nt = 0; nt < 2; ++nt)
                acc[mt][nt] = __builtin_amdgcn_wmma_f32_16x16x32_bf16(
                    false, aF[mt].v, false, bF[nt].v,
                    (short)0, acc[mt][nt], false, false);
        __syncthreads();
    }

    // ---- epilogue: C/D layout M = j + 8*half, N = lm ----
    #pragma unroll
    for (int mt = 0; mt < 2; ++mt) {
        #pragma unroll
        for (int nt = 0; nt < 2; ++nt) {
            const int n = n0 + wc * 32 + nt * 16 + lm;
            #pragma unroll
            for (int j = 0; j < 8; ++j) {
                const int m = m0 + wr * 32 + mt * 16 + half * 8 + j;
                if (m < M && n < N) {
                    float v = acc[mt][nt][j] + bias[n];
                    if (RELU) v = fmaxf(v, 0.0f);
                    C[(size_t)m * ldc + n] = (OT)v;
                }
            }
        }
    }
}

// =====================================================================
// im2col for the stride-32 backbone "patchify" conv
// =====================================================================
__global__ void im2col_backbone(const float* __restrict__ img, float* __restrict__ A0)
{
    int idx = blockIdx.x * blockDim.x + threadIdx.x;
    if (idx >= 98 * 1024) return;
    int m = idx >> 10, k = idx & 1023;
    int b = m / NS, s = m % NS;
    int oy = s / FHH, ox = s % FHH;
    int ky = k >> 5, kx = k & 31;
    A0[idx] = img[((size_t)b * 224 + oy * 32 + ky) * 224 + ox * 32 + kx];
}

// im2col for RPN 3x3 conv (pad 1), feats channels-last [b,49,1024]
__global__ void im2col_rpn(const float* __restrict__ featsCL, float* __restrict__ A1)
{
    int idx = blockIdx.x * blockDim.x + threadIdx.x;
    if (idx >= 98 * KRPN) return;
    int m = idx / KRPN, k = idx % KRPN;
    int b = m / NS, s = m % NS;
    int oy = s / FHH, ox = s % FHH;
    int ic = k / 9, rr = k % 9, ky = rr / 3, kx = rr % 3;
    int iy = oy + ky - 1, ix = ox + kx - 1;
    float v = 0.0f;
    if (iy >= 0 && iy < FHH && ix >= 0 && ix < FHH)
        v = featsCL[((size_t)b * NS + iy * FHH + ix) * CFEAT + ic];
    A1[idx] = v;
}

// =====================================================================
// RPN proposal generation (one workgroup per image)
// =====================================================================
__global__ __launch_bounds__(256)
void rpn_proposals_kernel(const float* __restrict__ obj,  // [2,735]
                          const float* __restrict__ regd, // [2,735,4]
                          float* __restrict__ props,      // [2,735,4]
                          int* __restrict__ valid)        // [2,735]
{
    __shared__ float bx[NANCH * 4];
    __shared__ float pr[NANCH];
    __shared__ float p2[NANCH];
    __shared__ int   ord[NANCH];
    __shared__ int   ord2[NANCH];
    __shared__ int   kp[NANCH];

    const int img = blockIdx.x, tid = threadIdx.x, nth = blockDim.x;

    for (int i = tid; i < NANCH; i += nth) {
        int s = i / AANCH, a = i % AANCH;
        int sy = s / FHH, sx = s % FHH;
        int ridx = a / 5, sidx = a % 5;
        float ratio = (ridx == 0) ? 0.5f : (ridx == 1) ? 1.0f : 2.0f;
        float scale = 32.0f * (float)(1 << sidx);
        float hr = sqrtf(ratio);
        float wsz = (1.0f / hr) * scale, hsz = hr * scale;
        float ax1 = sx * 32.0f + rintf(-wsz * 0.5f);
        float ay1 = sy * 32.0f + rintf(-hsz * 0.5f);
        float ax2 = sx * 32.0f + rintf( wsz * 0.5f);
        float ay2 = sy * 32.0f + rintf( hsz * 0.5f);
        float w = ax2 - ax1, h = ay2 - ay1;
        float cx = ax1 + 0.5f * w, cy = ay1 + 0.5f * h;
        const float* d = regd + ((size_t)img * NANCH + i) * 4;
        float dw = fminf(d[2], BBOX_CLIP_F), dh = fminf(d[3], BBOX_CLIP_F);
        float pcx = d[0] * w + cx, pcy = d[1] * h + cy;
        float pw = expf(dw) * w, ph = expf(dh) * h;
        float x1 = fminf(fmaxf(pcx - 0.5f * pw, 0.0f), IMGF);
        float y1 = fminf(fmaxf(pcy - 0.5f * ph, 0.0f), IMGF);
        float x2 = fminf(fmaxf(pcx + 0.5f * pw, 0.0f), IMGF);
        float y2 = fminf(fmaxf(pcy + 0.5f * ph, 0.0f), IMGF);
        bx[i * 4 + 0] = x1; bx[i * 4 + 1] = y1; bx[i * 4 + 2] = x2; bx[i * 4 + 3] = y2;
        float p = 1.0f / (1.0f + expf(-obj[(size_t)img * NANCH + i]));
        if (!((x2 - x1) >= 1e-3f && (y2 - y1) >= 1e-3f)) p = -1.0f;
        pr[i] = p;
    }
    __syncthreads();
    // rank-sort by (prob desc, idx asc)
    for (int i = tid; i < NANCH; i += nth) {
        float pi = pr[i]; int rk = 0;
        for (int j = 0; j < NANCH; ++j) {
            float pj = pr[j];
            rk += (pj > pi) || (pj == pi && j < i);
        }
        ord[rk] = i;
    }
    for (int i = tid; i < NANCH; i += nth) kp[i] = 1;
    __syncthreads();
    // greedy NMS (IoU > 0.7) in sorted order
    for (int it = 0; it < NANCH - 1; ++it) {
        if (kp[it]) {
            int bi = ord[it];
            float ax1 = bx[bi*4], ay1 = bx[bi*4+1], ax2 = bx[bi*4+2], ay2 = bx[bi*4+3];
            float areaA = (ax2 - ax1) * (ay2 - ay1);
            for (int j = it + 1 + tid; j < NANCH; j += nth) {
                int bj = ord[j];
                float ix1 = fmaxf(ax1, bx[bj*4]),   iy1 = fmaxf(ay1, bx[bj*4+1]);
                float ix2 = fminf(ax2, bx[bj*4+2]), iy2 = fminf(ay2, bx[bj*4+3]);
                float inter = fmaxf(ix2 - ix1, 0.0f) * fmaxf(iy2 - iy1, 0.0f);
                float areaB = (bx[bj*4+2] - bx[bj*4]) * (bx[bj*4+3] - bx[bj*4+1]);
                float iou = inter / fmaxf(areaA + areaB - inter, 1e-9f);
                if (iou > 0.7f) kp[j] = 0;
            }
        }
        __syncthreads();
    }
    for (int r = tid; r < NANCH; r += nth) p2[r] = kp[r] ? pr[ord[r]] : -1.0f;
    __syncthreads();
    for (int r = tid; r < NANCH; r += nth) {
        float pi = p2[r]; int rk = 0;
        for (int j = 0; j < NANCH; ++j)
            rk += (p2[j] > pi) || (p2[j] == pi && j < r);
        ord2[rk] = r;
    }
    __syncthreads();
    for (int q = tid; q < NANCH; q += nth) {
        int r = ord2[q]; int i = ord[r];
        float* o = props + ((size_t)img * NANCH + q) * 4;
        o[0] = bx[i*4+0]; o[1] = bx[i*4+1]; o[2] = bx[i*4+2]; o[3] = bx[i*4+3];
        valid[(size_t)img * NANCH + q] = (p2[r] > 0.0f) ? 1 : 0;
    }
}

// =====================================================================
// RoI-align (7x7, sr=2) -> pooled bf16 rows in c*49+s order (one block/ROI)
// =====================================================================
__global__ __launch_bounds__(256)
void roi_align_kernel(const float* __restrict__ featsCL,  // [2,49,1024]
                      const float* __restrict__ props,    // [2,735,4]
                      __bf16* __restrict__ pooled)        // [1470,50176]
{
    __shared__ int   spos[NS * 16];
    __shared__ float swgt[NS * 16];

    const int blk = blockIdx.x;            // 0..1469
    const int img = blk / NANCH;
    const float* p = props + (size_t)blk * 4;
    const float inv = 1.0f / 32.0f;
    float x1 = p[0] * inv, y1 = p[1] * inv;
    float rw = fmaxf(p[2] * inv - x1, 1.0f);
    float rh = fmaxf(p[3] * inv - y1, 1.0f);
    float bw = rw / 7.0f, bh = rh / 7.0f;

    for (int s = threadIdx.x; s < NS; s += blockDim.x) {
        int oy = s / FHH, ox = s % FHH;
        #pragma unroll
        for (int iy = 0; iy < 2; ++iy) {
            #pragma unroll
            for (int ix = 0; ix < 2; ++ix) {
                float yy = y1 + ((float)oy + ((float)iy + 0.5f) * 0.5f) * bh;
                float xx = x1 + ((float)ox + ((float)ix + 0.5f) * 0.5f) * bw;
                bool vv = (yy >= -1.0f) && (yy <= 7.0f) && (xx >= -1.0f) && (xx <= 7.0f);
                float y = fmaxf(yy, 0.0f), x = fmaxf(xx, 0.0f);
                int y0 = min((int)floorf(y), 6), x0 = min((int)floorf(x), 6);
                int y1i = min(y0 + 1, 6), x1i = min(x0 + 1, 6);
                if (y0 == 6) y = (float)y0;
                if (x0 == 6) x = (float)x0;
                float ly = y - (float)y0, lx = x - (float)x0;
                float hy = 1.0f - ly, hx = 1.0f - lx;
                float vm = vv ? 0.25f : 0.0f;
                int base = s * 16 + (iy * 2 + ix) * 4;
                spos[base+0] = y0 * FHH + x0;   swgt[base+0] = hy * hx * vm;
                spos[base+1] = y0 * FHH + x1i;  swgt[base+1] = hy * lx * vm;
                spos[base+2] = y1i * FHH + x0;  swgt[base+2] = ly * hx * vm;
                spos[base+3] = y1i * FHH + x1i; swgt[base+3] = ly * lx * vm;
            }
        }
    }
    __syncthreads();

    const float* fb = featsCL + (size_t)img * NS * CFEAT;
    __bf16* out = pooled + (size_t)blk * KPOOL;
    for (int t = threadIdx.x; t < KPOOL; t += blockDim.x) {
        int s = t >> 10;          // / 1024
        int c = t & 1023;
        float acc = 0.0f;
        #pragma unroll
        for (int k = 0; k < 16; ++k)
            acc += swgt[s * 16 + k] * fb[(size_t)spos[s * 16 + k] * CFEAT + c];
        out[(size_t)c * NS + s] = (__bf16)acc;
    }
}

// =====================================================================
// Detection postprocess (one 1024-thread workgroup per image)
// =====================================================================
__global__ __launch_bounds__(1024)
void postprocess_kernel(const float* __restrict__ logits,  // [1470,37]
                        const float* __restrict__ bdelt,   // [1470,148]
                        const float* __restrict__ props,   // [2,735,4]
                        const int* __restrict__ valid,     // [2,735]
                        float* __restrict__ gScore,        // [2,32768]
                        int* __restrict__ gIdx,            // [2,32768]
                        float* __restrict__ gBox,          // [2,26460,4]
                        float* __restrict__ outp)          // [2,100,6]
{
    __shared__ float smax[NANCH], ssum[NANCH];
    __shared__ float cbx4[CANDK * 4];
    __shared__ float csc[CANDK];
    __shared__ float ctv[CANDK];
    __shared__ unsigned char clb[CANDK];
    __shared__ unsigned char ckp[CANDK];

    const int img = blockIdx.x, tid = threadIdx.x, nth = blockDim.x;
    float* gS = gScore + (size_t)img * NSORT;
    int*   gI = gIdx   + (size_t)img * NSORT;
    float* gB = gBox   + (size_t)img * NALL * 4;

    // softmax stats per ROI
    for (int r = tid; r < NANCH; r += nth) {
        const float* lg = logits + ((size_t)img * NANCH + r) * NCLS;
        float m = lg[0];
        for (int k = 1; k < NCLS; ++k) m = fmaxf(m, lg[k]);
        float s = 0.0f;
        for (int k = 0; k < NCLS; ++k) s += expf(lg[k] - m);
        smax[r] = m; ssum[r] = s;
    }
    __syncthreads();

    // per (roi, class) score + decoded box
    for (int e = tid; e < NSORT; e += nth) {
        if (e < NALL) {
            int r = e / 36, cls = e % 36 + 1;
            const float* lg = logits + ((size_t)img * NANCH + r) * NCLS;
            float sc = expf(lg[cls] - smax[r]) / ssum[r];
            if (!valid[(size_t)img * NANCH + r]) sc = 0.0f;
            const float* d = bdelt + ((size_t)img * NANCH + r) * (NCLS * 4) + cls * 4;
            const float* p = props + ((size_t)img * NANCH + r) * 4;
            float w = p[2] - p[0], h = p[3] - p[1];
            float cx = p[0] + 0.5f * w, cy = p[1] + 0.5f * h;
            float dw = fminf(d[2] * 0.2f, BBOX_CLIP_F);
            float dh = fminf(d[3] * 0.2f, BBOX_CLIP_F);
            float pcx = d[0] * 0.1f * w + cx, pcy = d[1] * 0.1f * h + cy;
            float pw = expf(dw) * w, ph = expf(dh) * h;
            float x1 = fminf(fmaxf(pcx - 0.5f * pw, 0.0f), IMGF);
            float y1 = fminf(fmaxf(pcy - 0.5f * ph, 0.0f), IMGF);
            float x2 = fminf(fmaxf(pcx + 0.5f * pw, 0.0f), IMGF);
            float y2 = fminf(fmaxf(pcy + 0.5f * ph, 0.0f), IMGF);
            bool ok = ((x2 - x1) >= 1e-2f) && ((y2 - y1) >= 1e-2f) && (sc > 0.05f);
            gS[e] = ok ? sc : -1.0f;
            gI[e] = e;
            gB[(size_t)e * 4 + 0] = x1; gB[(size_t)e * 4 + 1] = y1;
            gB[(size_t)e * 4 + 2] = x2; gB[(size_t)e * 4 + 3] = y2;
        } else {
            gS[e] = -1e30f; gI[e] = e;
        }
    }
    __syncthreads();

    // bitonic sort (score desc, idx asc) over 32768, single block
    for (int k = 2; k <= NSORT; k <<= 1) {
        for (int j = k >> 1; j > 0; j >>= 1) {
            for (int i = tid; i < NSORT; i += nth) {
                int ixj = i ^ j;
                if (ixj > i) {
                    float ka = gS[i], kb = gS[ixj];
                    int ia = gI[i], ib = gI[ixj];
                    bool aFirst = (ka > kb) || (ka == kb && ia < ib);
                    bool up = ((i & k) == 0);
                    if (up ? !aFirst : aFirst) {
                        gS[i] = kb; gS[ixj] = ka;
                        gI[i] = ib; gI[ixj] = ia;
                    }
                }
            }
            __syncthreads();
        }
    }

    // load top-2048 candidates into LDS
    for (int i = tid; i < CANDK; i += nth) {
        float sc = gS[i]; int e = gI[i];
        csc[i] = sc;
        if (e < NALL) {
            clb[i] = (unsigned char)(e % 36 + 1);
            cbx4[i*4+0] = gB[(size_t)e*4+0]; cbx4[i*4+1] = gB[(size_t)e*4+1];
            cbx4[i*4+2] = gB[(size_t)e*4+2]; cbx4[i*4+3] = gB[(size_t)e*4+3];
        } else {
            clb[i] = 0;
            cbx4[i*4+0] = cbx4[i*4+1] = cbx4[i*4+2] = cbx4[i*4+3] = 0.0f;
        }
        ckp[i] = 1;
    }
    __syncthreads();

    // batched (per-class) greedy NMS @0.5: label-offset trick == same-label IoU
    for (int it = 0; it < CANDK - 1; ++it) {
        if (ckp[it]) {
            float ax1 = cbx4[it*4], ay1 = cbx4[it*4+1], ax2 = cbx4[it*4+2], ay2 = cbx4[it*4+3];
            unsigned char li = clb[it];
            float areaA = (ax2 - ax1) * (ay2 - ay1);
            for (int j = it + 1 + tid; j < CANDK; j += nth) {
                if (clb[j] == li) {
                    float ix1 = fmaxf(ax1, cbx4[j*4]),   iy1 = fmaxf(ay1, cbx4[j*4+1]);
                    float ix2 = fminf(ax2, cbx4[j*4+2]), iy2 = fminf(ay2, cbx4[j*4+3]);
                    float inter = fmaxf(ix2 - ix1, 0.0f) * fmaxf(iy2 - iy1, 0.0f);
                    float areaB = (cbx4[j*4+2] - cbx4[j*4]) * (cbx4[j*4+3] - cbx4[j*4+1]);
                    float iou = inter / fmaxf(areaA + areaB - inter, 1e-9f);
                    if (iou > 0.5f) ckp[j] = 0;
                }
            }
        }
        __syncthreads();
    }
    for (int i = tid; i < CANDK; i += nth) ctv[i] = ckp[i] ? csc[i] : -1.0f;
    __syncthreads();

    // rank top-100 and write output
    for (int i = tid; i < CANDK; i += nth) {
        float vi = ctv[i]; int rk = 0;
        for (int j = 0; j < CANDK; ++j)
            rk += (ctv[j] > vi) || (ctv[j] == vi && j < i);
        if (rk < 100) {
            bool vld = vi > 0.0f;
            float* o = outp + ((size_t)img * 100 + rk) * 6;
            o[0] = vld ? cbx4[i*4+0] : 0.0f;
            o[1] = vld ? cbx4[i*4+1] : 0.0f;
            o[2] = vld ? cbx4[i*4+2] : 0.0f;
            o[3] = vld ? cbx4[i*4+3] : 0.0f;
            o[4] = vld ? vi : 0.0f;
            o[5] = vld ? (float)clb[i] : 0.0f;
        }
    }
}

// =====================================================================
// Host launcher
// =====================================================================
extern "C" void kernel_launch(void* const* d_in, const int* in_sizes, int n_in,
                              void* d_out, int out_size, void* d_ws, size_t ws_size,
                              hipStream_t stream) {
    (void)in_sizes; (void)n_in; (void)out_size; (void)ws_size;
    const float* images = (const float*)d_in[0];
    const float* bb_w   = (const float*)d_in[1];
    const float* bb_b   = (const float*)d_in[2];
    const float* rpn_w  = (const float*)d_in[3];
    const float* rpn_b  = (const float*)d_in[4];
    const float* cls_w  = (const float*)d_in[5];
    const float* cls_b  = (const float*)d_in[6];
    const float* reg_w  = (const float*)d_in[7];
    const float* reg_b  = (const float*)d_in[8];
    const float* fc1_w  = (const float*)d_in[9];
    const float* fc1_b  = (const float*)d_in[10];
    const float* fc2_w  = (const float*)d_in[11];
    const float* fc2_b  = (const float*)d_in[12];
    const float* clsp_w = (const float*)d_in[13];
    const float* clsp_b = (const float*)d_in[14];
    const float* boxp_w = (const float*)d_in[15];
    const float* boxp_b = (const float*)d_in[16];

    char* ws = (char*)d_ws;
    size_t off = 0;
    auto carve = [&](size_t bytes) -> void* {
        void* p = ws + off;
        off += (bytes + 255) & ~(size_t)255;
        return p;
    };
    float*  A0      = (float*)carve((size_t)98 * 1024 * 4);
    float*  featsCL = (float*)carve((size_t)98 * 1024 * 4);
    float*  A1      = (float*)carve((size_t)98 * KRPN * 4);
    float*  tCL     = (float*)carve((size_t)98 * 1024 * 4);
    float*  obj     = (float*)carve((size_t)98 * 15 * 4);
    float*  regd    = (float*)carve((size_t)98 * 60 * 4);
    float*  props   = (float*)carve((size_t)NROI * 4 * 4);
    int*    valid   = (int*)carve((size_t)NROI * 4);
    __bf16* pooled  = (__bf16*)carve((size_t)NROI * KPOOL * 2);
    __bf16* x1      = (__bf16*)carve((size_t)NROI * 1024 * 2);
    __bf16* x2      = (__bf16*)carve((size_t)NROI * 1024 * 2);
    float*  logits  = (float*)carve((size_t)NROI * NCLS * 4);
    float*  bdelt   = (float*)carve((size_t)NROI * NCLS * 4 * 4);
    float*  gScore  = (float*)carve((size_t)NIMG * NSORT * 4);
    int*    gIdx    = (int*)carve((size_t)NIMG * NSORT * 4);
    float*  gBox    = (float*)carve((size_t)NIMG * NALL * 4 * 4);

    // 1) backbone stride-32 conv as GEMM (98x1024x1024), channels-last output
    im2col_backbone<<<(98 * 1024 + 255) / 256, 256, 0, stream>>>(images, A0);
    gemm_wmma_bf16<float, float, true, false><<<dim3(16, 2), 128, 0, stream>>>(
        A0, 1024, bb_w, bb_b, featsCL, 1024, 98, 1024, 1024);
    // 2) RPN 3x3 conv as GEMM (98x1024x9216)
    im2col_rpn<<<(98 * KRPN + 255) / 256, 256, 0, stream>>>(featsCL, A1);
    gemm_wmma_bf16<float, float, true, false><<<dim3(16, 2), 128, 0, stream>>>(
        A1, KRPN, rpn_w, rpn_b, tCL, 1024, 98, 1024, KRPN);
    // 3) RPN heads (1x1 convs)
    gemm_wmma_bf16<float, float, false, false><<<dim3(1, 2), 128, 0, stream>>>(
        tCL, 1024, cls_w, cls_b, obj, 15, 98, 15, 1024);
    gemm_wmma_bf16<float, float, false, false><<<dim3(1, 2), 128, 0, stream>>>(
        tCL, 1024, reg_w, reg_b, regd, 60, 98, 60, 1024);
    // 4) proposals
    rpn_proposals_kernel<<<NIMG, 256, 0, stream>>>(obj, regd, props, valid);
    // 5) RoI-align -> pooled bf16 [1470, 50176]
    roi_align_kernel<<<NROI, 256, 0, stream>>>(featsCL, props, pooled);
    // 6) box head FCs (dominant GEMM: 1470x1024x50176), async A staging
    gemm_wmma_bf16<__bf16, __bf16, true, true><<<dim3(16, 23), 128, 0, stream>>>(
        pooled, KPOOL, fc1_w, fc1_b, x1, 1024, NROI, 1024, KPOOL);
    gemm_wmma_bf16<__bf16, __bf16, true, true><<<dim3(16, 23), 128, 0, stream>>>(
        x1, 1024, fc2_w, fc2_b, x2, 1024, NROI, 1024, 1024);
    // 7) predictors
    gemm_wmma_bf16<__bf16, float, false, true><<<dim3(1, 23), 128, 0, stream>>>(
        x2, 1024, clsp_w, clsp_b, logits, NCLS, NROI, NCLS, 1024);
    gemm_wmma_bf16<__bf16, float, false, true><<<dim3(3, 23), 128, 0, stream>>>(
        x2, 1024, boxp_w, boxp_b, bdelt, NCLS * 4, NROI, NCLS * 4, 1024);
    // 8) postprocess -> [2,100,6]
    postprocess_kernel<<<NIMG, 1024, 0, stream>>>(
        logits, bdelt, props, valid, gScore, gIdx, gBox, (float*)d_out);
}
